// SNN_32186484916880
// MI455X (gfx1250) — matrix operand order repaired and checked by reading, not compiled, not run on hardware
//
#include <hip/hip_runtime.h>

typedef _Float16 v16h  __attribute__((ext_vector_type(16)));
typedef _Float16 half8 __attribute__((ext_vector_type(8)));
typedef float    v8f   __attribute__((ext_vector_type(8)));

#define BATCH   256
#define TSTEPS  100
#define IN_DIM  700
#define KPAD0   704                 // IN_DIM padded to multiple of 32
#define HID     1024
#define OUT_DIM 20
#define MROWS   (BATCH * TSTEPS)    // 25600

// ---------------------------------------------------------------------------
// x [B,T,IN] f32  ->  xh [T*B, KPAD0] f16 (permuted to GEMM row order,
// zero-padded K).
// ---------------------------------------------------------------------------
__global__ __launch_bounds__(256) void snn_cvt_x(
    const float* __restrict__ x, _Float16* __restrict__ xh)
{
    const long i  = (long)blockIdx.x * 256 + threadIdx.x;   // chunk-of-4 index
    const long r  = i / (KPAD0 / 4);                        // GEMM row t*B+b
    const int  k4 = (int)(i % (KPAD0 / 4)) * 4;
    const int  t  = (int)(r / BATCH);
    const int  b  = (int)(r % BATCH);
    const float* src = x + ((long)b * TSTEPS + t) * IN_DIM + k4;
    _Float16*    dst = xh + r * KPAD0 + k4;
#pragma unroll
    for (int j = 0; j < 4; ++j) {
        float v = 0.0f;
        if (k4 + j < IN_DIM) v = src[j];
        dst[j] = (_Float16)v;
    }
}

// W [N,K] f32 -> Wh [N,KP] f16 (zero-padded K).
__global__ __launch_bounds__(256) void snn_cvt_w(
    const float* __restrict__ W, _Float16* __restrict__ Wh,
    int N, int K, int KP)
{
    const long i = (long)blockIdx.x * 256 + threadIdx.x;
    if (i >= (long)N * (KP / 4)) return;
    const long n  = i / (KP / 4);
    const int  k4 = (int)(i % (KP / 4)) * 4;
    const float* src = W + n * K + k4;
    _Float16*    dst = Wh + n * KP + k4;
#pragma unroll
    for (int j = 0; j < 4; ++j) {
        float v = 0.0f;
        if (k4 + j < K) v = src[j];
        dst[j] = (_Float16)v;
    }
}

// ---------------------------------------------------------------------------
// f16 WMMA GEMM:  C[M x N] = A[M x K]_f16 * Wh^T   (Wh is [N x ldb] f16)
// 128 threads = 4 waves; block tile 128x128; wave tile 64x64 (4x4 accums).
// Register-pipelined global loads + LDS DOUBLE BUFFERING: one barrier per
// slab, ds_stores target the idle buffer after the WMMAs.  K must be a
// multiple of 64 slabs-wise (NS even: 22 for K=704, 32 for K=1024).
// ---------------------------------------------------------------------------
template <bool NGUARD>
__global__ __launch_bounds__(128) void snn_gemm_f16(
    const _Float16* __restrict__ A, int lda,
    const _Float16* __restrict__ B, int ldb,
    int K, int N,
    float* __restrict__ C, int ldc)
{
    __shared__ __align__(16) _Float16 lA[2][128][48];  // 32 used cols, 96B row
    __shared__ __align__(16) _Float16 lB[2][128][48];

    const int  tid    = threadIdx.x;
    const int  lane   = tid & 31;
    const int  wave   = tid >> 5;
    const long m_base = (long)blockIdx.x * 128;
    const long n_base = (long)blockIdx.y * 128;
    const int  wm     = (wave >> 1) * 64;
    const int  wn     = (wave & 1) * 64;

    v8f acc[4][4];
#pragma unroll
    for (int i = 0; i < 4; ++i)
#pragma unroll
        for (int j = 0; j < 4; ++j)
#pragma unroll
            for (int e = 0; e < 8; ++e) acc[i][j][e] = 0.0f;

    // ---- staging geometry: 4 chunks of 8 halves per thread per matrix ----
    const int  sr = tid >> 2;            // rows sr, sr+32, sr+64, sr+96
    const int  sc = (tid & 3) * 8;       // 8-half column chunk
    const long aStep = (long)32 * lda;
    const long bStep = (long)32 * ldb;
    const _Float16* gA0 = A + (m_base + sr) * lda + sc;
    const _Float16* gB0 = B + (n_base + sr) * ldb + sc;
    bool bok[4];
#pragma unroll
    for (int c = 0; c < 4; ++c)
        bok[c] = !NGUARD || (n_base + sr + 32 * c < N);

    half8 ra[4], rb[4];
    auto loadSlab = [&](int kofs) {      // global -> regs, no wait
#pragma unroll
        for (int c = 0; c < 4; ++c) {
            ra[c] = *reinterpret_cast<const half8*>(gA0 + c * aStep + kofs);
            if (bok[c]) {
                rb[c] = *reinterpret_cast<const half8*>(gB0 + c * bStep + kofs);
            } else {
#pragma unroll
                for (int j = 0; j < 8; ++j) rb[c][j] = (_Float16)0.0f;
            }
        }
    };
    auto storeSlab = [&](int bi) {       // regs -> LDS buffer bi
#pragma unroll
        for (int c = 0; c < 4; ++c) {
            *reinterpret_cast<half8*>(&lA[bi][sr + 32 * c][sc]) = ra[c];
            *reinterpret_cast<half8*>(&lB[bi][sr + 32 * c][sc]) = rb[c];
        }
    };

    const int kAo = (lane < 16) ? 0 : 16;   // A fragment K origin (contiguous layout)
    const int kBo = (lane < 16) ? 0 : 16;   // B fragment K origin
    const int fr  = lane & 15;

    auto compute = [&](int bi) {            // 16 WMMAs from LDS buffer bi
        v16h bf[4];
#pragma unroll
        for (int sj = 0; sj < 4; ++sj) {
            const int nrow = wn + sj * 16 + fr;
            half8 blo = *reinterpret_cast<const half8*>(&lB[bi][nrow][kBo]);
            half8 bhi = *reinterpret_cast<const half8*>(&lB[bi][nrow][kBo + 8]);
#pragma unroll
            for (int i = 0; i < 8; ++i) { bf[sj][i] = blo[i]; bf[sj][8 + i] = bhi[i]; }
        }
#pragma unroll
        for (int si = 0; si < 4; ++si) {
            const int mrow = wm + si * 16 + fr;
            half8 alo = *reinterpret_cast<const half8*>(&lA[bi][mrow][kAo]);
            half8 ahi = *reinterpret_cast<const half8*>(&lA[bi][mrow][kAo + 8]);
            v16h a;
#pragma unroll
            for (int i = 0; i < 8; ++i) { a[i] = alo[i]; a[8 + i] = ahi[i]; }
#pragma unroll
            for (int sj = 0; sj < 4; ++sj)
                acc[si][sj] = __builtin_amdgcn_wmma_f32_16x16x32_f16(
                    false, a, false, bf[sj], (short)0, acc[si][sj], false, false);
        }
    };

    const int NS = K >> 5;                  // even for all our shapes
    // ---- prologue: slab 0 into buffer 0 ----
    loadSlab(0);
    storeSlab(0);
    __syncthreads();

    // ---- steady state: 2 slabs per trip, no guards, 1 barrier per slab ----
    int s = 0;
    for (; s + 2 < NS; s += 2) {
        loadSlab((s + 1) * 32);             // issue next, no wait
        compute(0);                         // slab s from buf0 (immediately)
        storeSlab(1);                       // waits loads, fills idle buf1
        __syncthreads();
        loadSlab((s + 2) * 32);
        compute(1);                         // slab s+1 from buf1
        storeSlab(0);
        __syncthreads();
    }
    // ---- epilogue: slabs NS-2 (buf0), NS-1 ----
    loadSlab((s + 1) * 32);
    compute(0);
    storeSlab(1);
    __syncthreads();
    compute(1);

    // ---- store C (f32) ----
    const int ncol = lane & 15;
    const int moff = (lane < 16) ? 0 : 8;
#pragma unroll
    for (int si = 0; si < 4; ++si)
#pragma unroll
        for (int sj = 0; sj < 4; ++sj) {
            const long gm0 = m_base + wm + si * 16 + moff;
            const long gn  = n_base + wn + sj * 16 + ncol;
            if (!NGUARD || gn < N) {
#pragma unroll
                for (int r = 0; r < 8; ++r)
                    C[(gm0 + r) * ldc + gn] = acc[si][sj][r];
            }
        }
}

// ---------------------------------------------------------------------------
// Leaky membrane scans. reset uses pre-update membrane (snntorch subtract).
// Hidden layers emit f16 spikes ({0,1} exact) directly as next GEMM's A.
// ---------------------------------------------------------------------------
__global__ __launch_bounds__(256) void snn_scan_hidden(
    const float* __restrict__ cur, _Float16* __restrict__ spk,
    const float* __restrict__ thr_p)
{
    const long idx    = (long)blockIdx.x * blockDim.x + threadIdx.x;
    const float thr   = thr_p[0];
    const long stride = (long)BATCH * HID;
    float m = 0.0f;
    long off = idx;
    for (int t = 0; t < TSTEPS; ++t, off += stride) {
        float cu    = cur[off];
        float reset = (m > thr) ? 1.0f : 0.0f;
        m = 0.5f * m + cu - reset * thr;               // BETA = 0.5
        spk[off] = ((m - thr) > 0.0f) ? (_Float16)1.0f : (_Float16)0.0f;
    }
}

__global__ __launch_bounds__(256) void snn_scan_out(
    const float* __restrict__ cur, float* __restrict__ out,
    const float* __restrict__ thr_p)
{
    const long idx    = (long)blockIdx.x * blockDim.x + threadIdx.x;
    const float thr   = thr_p[0];
    const long stride = (long)BATCH * OUT_DIM;
    float* spk_rec = out;
    float* mem_rec = out + (long)TSTEPS * BATCH * OUT_DIM;
    float m = 0.0f;
    long off = idx;
    for (int t = 0; t < TSTEPS; ++t, off += stride) {
        float cu    = cur[off];
        float reset = (m > thr) ? 1.0f : 0.0f;
        m = 0.5f * m + cu - reset * thr;
        spk_rec[off] = ((m - thr) > 0.0f) ? 1.0f : 0.0f;
        mem_rec[off] = m;
    }
}

// ---------------------------------------------------------------------------
extern "C" void kernel_launch(void* const* d_in, const int* in_sizes, int n_in,
                              void* d_out, int out_size, void* d_ws, size_t ws_size,
                              hipStream_t stream) {
    const float* x    = (const float*)d_in[0];   // [B, T, IN]
    const float* W0   = (const float*)d_in[1];   // [HID, IN]
    const float* W1   = (const float*)d_in[2];   // [HID, HID]
    const float* W2   = (const float*)d_in[3];   // [OUT, HID]
    const float* thr0 = (const float*)d_in[4];
    const float* thr1 = (const float*)d_in[5];
    const float* thr2 = (const float*)d_in[6];

    // workspace carve-up (f32 first to keep 16B alignment everywhere)
    float*     curb = (float*)d_ws;                              // [MROWS,HID]
    float*     cur2 = curb + (size_t)MROWS * HID;                // [MROWS,OUT]
    _Float16*  spkb = (_Float16*)(cur2 + (size_t)MROWS * OUT_DIM);
    _Float16*  xh   = spkb + (size_t)MROWS * HID;                // [MROWS,KPAD0]
    _Float16*  w0h  = xh   + (size_t)MROWS * KPAD0;              // [HID,KPAD0]
    _Float16*  w1h  = w0h  + (size_t)HID * KPAD0;                // [HID,HID]
    _Float16*  w2h  = w1h  + (size_t)HID * HID;                  // [OUT,HID]
    float*     out  = (float*)d_out;

    // ---- one-time f16 conversion passes (fold x permute + K padding) ----
    snn_cvt_x<<<(MROWS * (KPAD0 / 4)) / 256, 256, 0, stream>>>(x, xh);
    snn_cvt_w<<<(HID * (KPAD0 / 4) + 255) / 256, 256, 0, stream>>>(W0, w0h, HID, IN_DIM, KPAD0);
    snn_cvt_w<<<(HID * (HID / 4) + 255) / 256, 256, 0, stream>>>(W1, w1h, HID, HID, HID);
    snn_cvt_w<<<(OUT_DIM * (HID / 4) + 255) / 256, 256, 0, stream>>>(W2, w2h, OUT_DIM, HID, HID);

    dim3 blk(128);
    dim3 gH(MROWS / 128, HID / 128);   // 200 x 8
    dim3 gO(MROWS / 128, 1);

    // Layer 0: cur0 = xh @ W0^T   (K padded to 704, zero tail contributes 0)
    snn_gemm_f16<false><<<gH, blk, 0, stream>>>(xh, KPAD0, w0h, KPAD0, KPAD0, HID, curb, HID);
    snn_scan_hidden<<<(BATCH * HID) / 256, 256, 0, stream>>>(curb, spkb, thr0);
    // Layer 1: cur1 = spk0 @ W1^T
    snn_gemm_f16<false><<<gH, blk, 0, stream>>>(spkb, HID, w1h, HID, HID, HID, curb, HID);
    snn_scan_hidden<<<(BATCH * HID) / 256, 256, 0, stream>>>(curb, spkb, thr1);
    // Layer 2: cur2 = spk1 @ W2^T  (N=20 guarded)
    snn_gemm_f16<true><<<gO, blk, 0, stream>>>(spkb, HID, w2h, HID, HID, OUT_DIM, cur2, OUT_DIM);
    snn_scan_out<<<(BATCH * OUT_DIM) / 256, 256, 0, stream>>>(cur2, out, thr2);
}
